// AttentionBlock_24824910970983
// MI455X (gfx1250) — matrix operand order repaired and checked by reading, not compiled, not run on hardware
//
#include <hip/hip_runtime.h>
#include <hip/hip_bf16.h>
#include <math.h>

// ---------------------------------------------------------------------------
// AttentionBlock for MI455X (gfx1250, wave32, WMMA).
//   B=512, K=8, C=64, E=8 -> S=E^3=512, NIN=32768, HID=128, CF=32.
// Pipeline (all on `stream`):
//   1) prep_w0:    W0 (32768x128) -> hi/lo f16 per-lane B-fragment layout
//   2) layer1:     H = LeakyReLU(A @ W0 + b0) via split-f16 WMMA (f32-accurate)
//                  A tiles staged with GLOBAL_LOAD_ASYNC_TO_LDS_B128 (ASYNCcnt),
//                  double-buffered so HBM fetch overlaps the WMMA pipeline.
//   3) mlp_tail:   3 small 128-wide layers (VALU; negligible FLOPs)
//   4) scores:     l2-normalize, dot, sharp softmax, switch gate (wave32 shfl)
//   5) prep2:      M = o_w@g_w, bias2 = o_w@g_b + o_b   (conv fusion)
//   6) attn_out:   pbar = sum_k w_k p[b,k]; y = M@pbar; gated blend with x
// ---------------------------------------------------------------------------

typedef __attribute__((ext_vector_type(16))) _Float16 v16h;
typedef __attribute__((ext_vector_type(8)))  _Float16 v8h;
typedef __attribute__((ext_vector_type(8)))  float    v8f;

#define B_   512
#define K_   8
#define C_   64
#define S_   512          // E^3
#define NIN_ 32768
#define HID_ 128
#define CF_  32
#define LEAK_ 0.01f

// ---------------------------------------------------------------------------
// 1) Pre-swizzle W0 (NIN x 128, row-major) into WMMA B-fragment order.
// B 32x16 f16 layout: lane<16 -> K=0..7,16..23 (N=lane); lane>=16 -> K=8..15,24..31.
// Element e of the per-lane v16h: k_local = (e<8 ? e : e+8) + (lane<16 ? 0 : 8).
// Storage: [ktile(1024)][ntile(8)][lane(32)][e(16)] halves, hi and lo arrays.
// ---------------------------------------------------------------------------
__global__ void prep_w0_kernel(const float* __restrict__ W,
                               _Float16* __restrict__ Whi,
                               _Float16* __restrict__ Wlo) {
  const int kt   = blockIdx.x;           // 0..1023
  const int nt   = blockIdx.y;           // 0..7
  const int lane = threadIdx.x;          // 0..31
  const int n    = nt * 16 + (lane & 15);
  const int kb   = kt * 32 + ((lane < 16) ? 0 : 8);
  const size_t base = ((size_t)(kt * 8 + nt) * 32 + lane) * 16;
#pragma unroll
  for (int e = 0; e < 16; ++e) {
    float w = W[(size_t)(kb + (e < 8 ? e : e + 8)) * HID_ + n];
    _Float16 h = (_Float16)w;
    Whi[base + e] = h;
    Wlo[base + e] = (_Float16)(w - (float)h);
  }
}

// ---------------------------------------------------------------------------
// 2) Layer-1 GEMM: H(R,128) = LeakyReLU(A(R,32768) @ W0 + b0)
// One workgroup = 16 output rows, 8 waves = 8 N-tiles of 16.
// Raw f32 A tiles (16x256) are fetched with async global->LDS copies into a
// ping-pong buffer (tile i+1 in flight while tile i is split + WMMA'd), then
// split to hi/lo f16 in LDS once; every wave builds A fragments with aligned
// v8h LDS reads.  Half stride 264: bank(row) = 4*row -> conflict-free.
// ---------------------------------------------------------------------------
#define KT_  256
#define AST_ 264

__device__ __forceinline__ void async_tile_load(const float* gsrc, unsigned ldsDst) {
  // 64B per thread: four async b128 copies, offset applied to both addresses.
  asm volatile("global_load_async_to_lds_b128 %0, %1, off"
               :: "v"(ldsDst), "v"(gsrc) : "memory");
  asm volatile("global_load_async_to_lds_b128 %0, %1, off offset:16"
               :: "v"(ldsDst), "v"(gsrc) : "memory");
  asm volatile("global_load_async_to_lds_b128 %0, %1, off offset:32"
               :: "v"(ldsDst), "v"(gsrc) : "memory");
  asm volatile("global_load_async_to_lds_b128 %0, %1, off offset:48"
               :: "v"(ldsDst), "v"(gsrc) : "memory");
}

__global__ __launch_bounds__(256) void layer1_kernel(
    const float* __restrict__ A, const _Float16* __restrict__ Whi,
    const _Float16* __restrict__ Wlo, const float* __restrict__ bias,
    float* __restrict__ H) {
  __shared__ __align__(16) float    sRaw[2][16 * KT_];   // ping-pong raw tiles
  __shared__ __align__(16) _Float16 sAhi[16 * AST_];
  __shared__ __align__(16) _Float16 sAlo[16 * AST_];
  const int t       = threadIdx.x;
  const int lane    = t & 31;
  const int nt      = t >> 5;              // wave id == N tile (0..7)
  const int rowbase = blockIdx.x * 16;
  const int rl      = t >> 4;              // staging row 0..15
  const int c0      = (t & 15) * 16;       // staging col start
  const float* Arow = A + (size_t)(rowbase + rl) * NIN_ + c0;
  // LDS byte offsets of this thread's staging region (generic ptr low 32 bits)
  const unsigned ldsA = (unsigned)(uintptr_t)&sRaw[0][rl * KT_ + c0];
  const unsigned ldsB = (unsigned)(uintptr_t)&sRaw[1][rl * KT_ + c0];

  v8f acc = {0.f, 0.f, 0.f, 0.f, 0.f, 0.f, 0.f, 0.f};

  const int row  = lane & 15;
  const int offA = (lane < 16) ? 0 : 8;    // A-frag K sub-offset (ISA 16x32 map)

  // prologue: start async fetch of tile 0 into buffer 0
  async_tile_load(Arow, ldsA);

  int buf = 0;
  for (int kt = 0; kt < NIN_; kt += KT_) {
    const bool hasNext = (kt + KT_) < NIN_;
    // prefetch next tile into the other buffer (recycled: its last reader
    // finished before the convert/wmma barrier of the previous iteration)
    if (hasNext) {
      async_tile_load(Arow + kt + KT_, buf ? ldsA : ldsB);
      asm volatile("s_wait_asynccnt 0x4" ::: "memory");  // tile kt complete
    } else {
      asm volatile("s_wait_asynccnt 0x0" ::: "memory");
    }
    __syncthreads();   // publish raw tile; halves free (prev wmma done)

    // ---- hi/lo split of this thread's own 16 floats ----
    {
      const float4* src = (const float4*)&sRaw[buf][rl * KT_ + c0];
#pragma unroll
      for (int q = 0; q < 4; ++q) {
        float4 f = src[q];
        int hb = rl * AST_ + c0 + q * 4;
        _Float16 h;
        h = (_Float16)f.x; sAhi[hb + 0] = h; sAlo[hb + 0] = (_Float16)(f.x - (float)h);
        h = (_Float16)f.y; sAhi[hb + 1] = h; sAlo[hb + 1] = (_Float16)(f.y - (float)h);
        h = (_Float16)f.z; sAhi[hb + 2] = h; sAlo[hb + 2] = (_Float16)(f.z - (float)h);
        h = (_Float16)f.w; sAhi[hb + 3] = h; sAlo[hb + 3] = (_Float16)(f.w - (float)h);
      }
    }
    __syncthreads();

    // ---- 8 k-steps of 32, 3 split WMMAs each ----
#pragma unroll
    for (int ko = 0; ko < KT_; ko += 32) {
      const int ktile = (kt + ko) >> 5;
      const size_t bidx = (size_t)(ktile * 8 + nt) * 32 + lane;
      v16h bh = *((const v16h*)Whi + bidx);
      v16h bl = *((const v16h*)Wlo + bidx);

      const int hbase = row * AST_ + ko + offA;
      v8h a0 = *(const v8h*)(sAhi + hbase);
      v8h a1 = *(const v8h*)(sAhi + hbase + 16);
      v8h l0 = *(const v8h*)(sAlo + hbase);
      v8h l1 = *(const v8h*)(sAlo + hbase + 16);
      v16h ah, al;
#pragma unroll
      for (int j = 0; j < 8; ++j) {
        ah[j] = a0[j]; ah[8 + j] = a1[j];
        al[j] = l0[j]; al[8 + j] = l1[j];
      }
      acc = __builtin_amdgcn_wmma_f32_16x16x32_f16(false, ah, false, bh, (short)0, acc, false, false);
      acc = __builtin_amdgcn_wmma_f32_16x16x32_f16(false, ah, false, bl, (short)0, acc, false, false);
      acc = __builtin_amdgcn_wmma_f32_16x16x32_f16(false, al, false, bh, (short)0, acc, false, false);
    }
    buf ^= 1;
  }

  // C/D 16x16 f32 layout: VGPR r -> M = r (lane<16) or r+8 (lane>=16), N = lane&15
  const int col = nt * 16 + (lane & 15);
  const float bc = bias[col];
  const int radd = (lane < 16) ? 0 : 8;
#pragma unroll
  for (int r = 0; r < 8; ++r) {
    float v = acc[r] + bc;
    v = (v > 0.0f) ? v : LEAK_ * v;
    H[(size_t)(rowbase + r + radd) * HID_ + col] = v;
  }
}

// ---------------------------------------------------------------------------
// 3) Tail layers: 128->128 (LReLU) -> 128->128 (LReLU) -> 128->32
// ---------------------------------------------------------------------------
__global__ __launch_bounds__(128) void mlp_tail_kernel(
    const float* __restrict__ H,
    const float* __restrict__ W1, const float* __restrict__ b1,
    const float* __restrict__ W2, const float* __restrict__ b2,
    const float* __restrict__ W3, const float* __restrict__ b3,
    float* __restrict__ F) {
  __shared__ float ha[HID_];
  __shared__ float hb[HID_];
  const int t = threadIdx.x;
  const size_t row = blockIdx.x;
  ha[t] = H[row * HID_ + t];
  __syncthreads();
  float acc = b1[t];
  for (int k = 0; k < HID_; ++k) acc = fmaf(ha[k], W1[k * HID_ + t], acc);
  acc = (acc > 0.0f) ? acc : LEAK_ * acc;
  hb[t] = acc;
  __syncthreads();
  acc = b2[t];
  for (int k = 0; k < HID_; ++k) acc = fmaf(hb[k], W2[k * HID_ + t], acc);
  acc = (acc > 0.0f) ? acc : LEAK_ * acc;
  ha[t] = acc;           // ha last read before previous barrier -> safe
  __syncthreads();
  if (t < CF_) {
    float a4 = b3[t];
    for (int k = 0; k < HID_; ++k) a4 = fmaf(ha[k], W3[k * CF_ + t], a4);
    F[row * CF_ + t] = a4;
  }
}

// ---------------------------------------------------------------------------
// 4) Scores: l2norm, dots, switch gate, sharpness softmax. One wave per b.
// ---------------------------------------------------------------------------
__global__ __launch_bounds__(32) void scores_kernel(
    const float* __restrict__ XF, const float* __restrict__ PF,
    float* __restrict__ WTS, float* __restrict__ SW) {
  const int b = blockIdx.x;
  const int lane = threadIdx.x;
  float xv = XF[(size_t)b * CF_ + lane];
  float ss = xv * xv;
  for (int o = 16; o > 0; o >>= 1) ss += __shfl_xor(ss, o, 32);
  xv /= fmaxf(sqrtf(ss), 1e-12f);
  float sc[K_];
#pragma unroll
  for (int k = 0; k < K_; ++k) {
    float pv = PF[((size_t)b * K_ + k) * CF_ + lane];
    float sp = pv * pv;
    for (int o = 16; o > 0; o >>= 1) sp += __shfl_xor(sp, o, 32);
    pv /= fmaxf(sqrtf(sp), 1e-12f);
    float d = xv * pv;
    for (int o = 16; o > 0; o >>= 1) d += __shfl_xor(d, o, 32);
    sc[k] = d;
  }
  float mx = sc[0];
#pragma unroll
  for (int k = 1; k < K_; ++k) mx = fmaxf(mx, sc[k]);
  const float sharp = (float)(CF_ * 8 * 8 * 8 * 4);   // 65536
  float es = 0.0f;
  float w[K_];
#pragma unroll
  for (int k = 0; k < K_; ++k) { w[k] = expf(sharp * (sc[k] - mx)); es += w[k]; }
  if (lane == 0) {
    SW[b] = fmaxf(mx, 0.0f);
#pragma unroll
    for (int k = 0; k < K_; ++k) WTS[(size_t)b * K_ + k] = w[k] / es;
  }
}

// ---------------------------------------------------------------------------
// 5) Fold the two 1x1 convs: M = o_w @ g_w, bias2 = o_w @ g_b + o_b
// ---------------------------------------------------------------------------
__global__ __launch_bounds__(256) void prep2_kernel(
    const float* __restrict__ gw, const float* __restrict__ gb,
    const float* __restrict__ ow, const float* __restrict__ ob,
    float* __restrict__ M, float* __restrict__ B2) {
  const int t = threadIdx.x;
  const int o = t >> 2;
  const int cb = (t & 3) * 16;
  for (int j = 0; j < 16; ++j) {
    const int c = cb + j;
    float acc = 0.0f;
    for (int cc = 0; cc < C_; ++cc) acc = fmaf(ow[o * C_ + cc], gw[cc * C_ + c], acc);
    M[o * C_ + c] = acc;
  }
  if (t < C_) {
    float acc = ob[t];
    for (int c = 0; c < C_; ++c) acc = fmaf(ow[t * C_ + c], gb[c], acc);
    B2[t] = acc;
  }
}

// ---------------------------------------------------------------------------
// 6) Fused output: pbar = sum_k w_k p[b,k]  (LDS, hi/lo split, transposed so
//    B fragments are two k-contiguous 8-half runs), y = M @ pbar via split WMMA,
//    out = x*(1-sw) + (y + bias2)*sw.  One workgroup per batch element.
// ---------------------------------------------------------------------------
#define MSTH_ 72   // half strides (pad for bank-conflict-free fragment reads)
#define PSTH_ 72

__global__ __launch_bounds__(256) void attn_out_kernel(
    const float* __restrict__ P, const float* __restrict__ X,
    const float* __restrict__ M, const float* __restrict__ B2,
    const float* __restrict__ WTS, const float* __restrict__ SW,
    float* __restrict__ OUT) {
  __shared__ __align__(16) _Float16 sMhi[C_ * MSTH_];
  __shared__ __align__(16) _Float16 sMlo[C_ * MSTH_];
  __shared__ __align__(16) _Float16 sphi[128 * PSTH_];   // [s][c]
  __shared__ __align__(16) _Float16 splo[128 * PSTH_];
  const int b = blockIdx.x;
  const int t = threadIdx.x;
  const int lane = t & 31;
  const int wave = t >> 5;

  for (int i = t; i < C_ * C_; i += 256) {
    const int o = i >> 6, c = i & 63;
    float v = M[i];
    _Float16 h = (_Float16)v;
    sMhi[o * MSTH_ + c] = h;
    sMlo[o * MSTH_ + c] = (_Float16)(v - (float)h);
  }
  float wk[K_];
#pragma unroll
  for (int k = 0; k < K_; ++k) wk[k] = WTS[(size_t)b * K_ + k];
  const float swv = SW[b];
  __syncthreads();

  const int row  = lane & 15;
  const int offA = (lane < 16) ? 0 : 8;
  const int offB = (lane < 16) ? 0 : 8;

  for (int s0 = 0; s0 < S_; s0 += 128) {
    // ---- weighted sum over K candidates (coalesced along s) ----
    for (int i = t; i < C_ * 128; i += 256) {
      const int c = i >> 7, ss = i & 127;
      const float* pp = P + ((size_t)b * K_ * C_ + c) * S_ + s0 + ss;
      float acc = 0.0f;
#pragma unroll
      for (int k = 0; k < K_; ++k) acc = fmaf(wk[k], pp[(size_t)k * (C_ * S_)], acc);
      _Float16 h = (_Float16)acc;
      sphi[ss * PSTH_ + c] = h;
      splo[ss * PSTH_ + c] = (_Float16)(acc - (float)h);
    }
    __syncthreads();

    // ---- y(64x128) = M(64x64) @ pbar(64x128): 32 tiles over 8 waves ----
    for (int tile = wave; tile < 32; tile += 8) {
      const int ot = tile >> 3, st = tile & 7;
      v8f acc = {0.f, 0.f, 0.f, 0.f, 0.f, 0.f, 0.f, 0.f};
#pragma unroll
      for (int kb = 0; kb < C_; kb += 32) {
        const int abase = (ot * 16 + row) * MSTH_ + kb + offA;
        v8h ah0 = *(const v8h*)(sMhi + abase);
        v8h ah1 = *(const v8h*)(sMhi + abase + 16);
        v8h al0 = *(const v8h*)(sMlo + abase);
        v8h al1 = *(const v8h*)(sMlo + abase + 16);
        v16h Ah, Al;
#pragma unroll
        for (int j = 0; j < 8; ++j) {
          Ah[j] = ah0[j]; Ah[8 + j] = ah1[j];
          Al[j] = al0[j]; Al[8 + j] = al1[j];
        }
        const int bbase = (st * 16 + row) * PSTH_ + kb + offB;
        v8h bh0 = *(const v8h*)(sphi + bbase);
        v8h bh1 = *(const v8h*)(sphi + bbase + 16);
        v8h bl0 = *(const v8h*)(splo + bbase);
        v8h bl1 = *(const v8h*)(splo + bbase + 16);
        v16h Bh, Bl;
#pragma unroll
        for (int j = 0; j < 8; ++j) {
          Bh[j] = bh0[j]; Bh[8 + j] = bh1[j];
          Bl[j] = bl0[j]; Bl[8 + j] = bl1[j];
        }
        acc = __builtin_amdgcn_wmma_f32_16x16x32_f16(false, Ah, false, Bh, (short)0, acc, false, false);
        acc = __builtin_amdgcn_wmma_f32_16x16x32_f16(false, Ah, false, Bl, (short)0, acc, false, false);
        acc = __builtin_amdgcn_wmma_f32_16x16x32_f16(false, Al, false, Bh, (short)0, acc, false, false);
      }
      const int sg = s0 + st * 16 + (lane & 15);
#pragma unroll
      for (int r = 0; r < 8; ++r) {
        const int o = ot * 16 + r + offA;
        const float v = acc[r] + B2[o];
        const size_t idx = ((size_t)b * C_ + o) * S_ + sg;
        OUT[idx] = X[idx] * (1.0f - swv) + v * swv;
      }
    }
    __syncthreads();
  }
}

// ---------------------------------------------------------------------------
// Host launcher
// ---------------------------------------------------------------------------
extern "C" void kernel_launch(void* const* d_in, const int* in_sizes, int n_in,
                              void* d_out, int out_size, void* d_ws, size_t ws_size,
                              hipStream_t stream) {
  (void)in_sizes; (void)out_size; (void)ws_size;
  const float *x, *p, *tw[4], *tb[4], *pw[4], *pb[4], *gw, *gb, *ow, *ob;
  if (n_in >= 22) {   // pytree leaves passed individually
    x = (const float*)d_in[0]; p = (const float*)d_in[1];
    for (int i = 0; i < 4; ++i) tw[i] = (const float*)d_in[2 + i];
    for (int i = 0; i < 4; ++i) tb[i] = (const float*)d_in[6 + i];
    for (int i = 0; i < 4; ++i) pw[i] = (const float*)d_in[10 + i];
    for (int i = 0; i < 4; ++i) pb[i] = (const float*)d_in[14 + i];
    gw = (const float*)d_in[18]; gb = (const float*)d_in[19];
    ow = (const float*)d_in[20]; ob = (const float*)d_in[21];
  } else {            // lists flattened/concatenated per reference argument
    x = (const float*)d_in[0]; p = (const float*)d_in[1];
    const float* twb = (const float*)d_in[2];
    const float* tbb = (const float*)d_in[3];
    const float* pwb = (const float*)d_in[4];
    const float* pbb = (const float*)d_in[5];
    tw[0] = twb; tw[1] = tw[0] + (size_t)NIN_ * HID_;
    tw[2] = tw[1] + HID_ * HID_; tw[3] = tw[2] + HID_ * HID_;
    tb[0] = tbb; tb[1] = tb[0] + HID_; tb[2] = tb[1] + HID_; tb[3] = tb[2] + HID_;
    pw[0] = pwb; pw[1] = pw[0] + (size_t)NIN_ * HID_;
    pw[2] = pw[1] + HID_ * HID_; pw[3] = pw[2] + HID_ * HID_;
    pb[0] = pbb; pb[1] = pb[0] + HID_; pb[2] = pb[1] + HID_; pb[3] = pb[2] + HID_;
    gw = (const float*)d_in[6]; gb = (const float*)d_in[7];
    ow = (const float*)d_in[8]; ob = (const float*)d_in[9];
  }

  // workspace carve-up (~36.6 MB)
  _Float16* WThi = (_Float16*)d_ws;
  _Float16* WTlo = WThi + (size_t)NIN_ * HID_;
  _Float16* WPhi = WTlo + (size_t)NIN_ * HID_;
  _Float16* WPlo = WPhi + (size_t)NIN_ * HID_;
  float* HX  = (float*)(WPlo + (size_t)NIN_ * HID_);
  float* HP  = HX + (size_t)B_ * HID_;
  float* XF  = HP + (size_t)B_ * K_ * HID_;
  float* PF  = XF + (size_t)B_ * CF_;
  float* WTS = PF + (size_t)B_ * K_ * CF_;
  float* SW  = WTS + (size_t)B_ * K_;
  float* MM  = SW + B_;
  float* B2  = MM + C_ * C_;
  float* out = (float*)d_out;

  prep_w0_kernel<<<dim3(NIN_ / 32, HID_ / 16), 32, 0, stream>>>(tw[0], WThi, WTlo);
  prep_w0_kernel<<<dim3(NIN_ / 32, HID_ / 16), 32, 0, stream>>>(pw[0], WPhi, WPlo);
  layer1_kernel<<<B_ / 16, 256, 0, stream>>>(x, WThi, WTlo, tb[0], HX);
  layer1_kernel<<<(B_ * K_) / 16, 256, 0, stream>>>(p, WPhi, WPlo, pb[0], HP);
  mlp_tail_kernel<<<B_, 128, 0, stream>>>(HX, tw[1], tb[1], tw[2], tb[2], tw[3], tb[3], XF);
  mlp_tail_kernel<<<B_ * K_, 128, 0, stream>>>(HP, pw[1], pb[1], pw[2], pb[2], pw[3], pb[3], PF);
  scores_kernel<<<B_, 32, 0, stream>>>(XF, PF, WTS, SW);
  prep2_kernel<<<1, 256, 0, stream>>>(gw, gb, ow, ob, MM, B2);
  attn_out_kernel<<<B_, 256, 0, stream>>>(p, x, MM, B2, WTS, SW, out);
}
